// MultiHeadAttention_39840116638479
// MI455X (gfx1250) — compile-verified, matrix operand order checked by reading, not run on hardware
//
#include <hip/hip_runtime.h>
#include <hip/hip_bf16.h>

// ---------------- types & helpers ----------------
typedef __attribute__((ext_vector_type(16))) __bf16 bf16x16;
typedef __attribute__((ext_vector_type(8)))  float  f32x8;
typedef unsigned short u16;

union FragA {
    bf16x16 v;
    uint4   q[2];
    u16     u[16];
};
static_assert(sizeof(FragA) == 32, "frag size");

__device__ __forceinline__ f32x8 zero8() {
    f32x8 c;
#pragma unroll
    for (int i = 0; i < 8; ++i) c[i] = 0.0f;
    return c;
}

__device__ __forceinline__ u16 f32_to_bf16(float f) {
    union { float f; unsigned u; } x{f};
    unsigned u = x.u;
    u += 0x7FFFu + ((u >> 16) & 1u);       // round-to-nearest-even
    return (u16)(u >> 16);
}

__device__ __forceinline__ f32x8 wmma_bf16(const FragA& a, const FragA& b, f32x8 c) {
    // D = A(16x32 bf16) * B(32x16 bf16) + C(16x16 f32)
    return __builtin_amdgcn_wmma_f32_16x16x32_bf16(
        false, a.v, false, b.v, (short)0, c, false, false);
}

__device__ __forceinline__ float redmax16(float v) {
#pragma unroll
    for (int off = 1; off <= 8; off <<= 1)
        v = fmaxf(v, __shfl_xor(v, off, 32));   // xor<=8 stays inside 16-lane half
    return v;
}
__device__ __forceinline__ float redsum16(float v) {
#pragma unroll
    for (int off = 1; off <= 8; off <<= 1)
        v += __shfl_xor(v, off, 32);
    return v;
}

// ---------------- problem constants ----------------
#define BB      2
#define SS      2048
#define DM      1024
#define NH      16
#define DK      64
#define MROWS   (BB * SS)          // 4096
#define NEGV    (-1e10f)
#define L2E     1.4426950408889634f
#define SCALE   (1.0f / 32.0f)     // 1/sqrt(D_MODEL)

// ---------------- f32 -> bf16 convert ----------------
__global__ void k_f2bf(const float* __restrict__ in, u16* __restrict__ out, int n) {
    int i = blockIdx.x * blockDim.x + threadIdx.x;
    if (i < n) out[i] = f32_to_bf16(in[i]);
}

// ---------------- tiled bf16 WMMA GEMM (128x64 tile, BK=64, double-buffered) ---
// C[M,N] = A[M,K] * B[K,N]; A,B bf16 row-major.
// mode 0: bf16 out [M,N]; mode 1: bf16 out transposed [N,M]; mode 2: f32 out [M,N]
// A tile is staged with CDNA5 async global->LDS DMA (ASYNCcnt); B is staged
// through registers because it is transposed into [n][k] for the WMMA B operand.
__global__ __launch_bounds__(256)
void k_gemm_bf16(const u16* __restrict__ A, const u16* __restrict__ B,
                 void* __restrict__ Cout, int M, int N, int K, int mode) {
    __shared__ __align__(16) u16 As[2][128][72];   // [m][k] padded (144B rows)
    __shared__ __align__(16) u16 Bst[2][64][72];   // [n][k] padded (transposed)

    const int t     = threadIdx.x;
    const int lane  = t & 31;
    const int wave  = t >> 5;
    const int gm0   = blockIdx.x * 128;
    const int gn0   = blockIdx.y * 64;
    const int mi2   = wave >> 1;            // 0..3 : 32-row block
    const int nj0   = (wave & 1) * 2;       // 2 subtile columns per wave
    const int row16 = lane & 15;
    const int half  = lane >> 4;
    const int kb8   = half * 8;

    f32x8 acc[2][2];
#pragma unroll
    for (int im = 0; im < 2; ++im)
#pragma unroll
        for (int in = 0; in < 2; ++in) acc[im][in] = zero8();

    // ---- staging helpers ----
    auto stageA_async = [&](int k0, int buf) {
#pragma unroll
        for (int j = 0; j < 4; ++j) {               // 128x64 u16 = 1024 x 16B chunks
            const int id = t + j * 256;
            const int ar = id >> 3, ac = (id & 7) * 8;
            const u16* gp = A + (size_t)(gm0 + ar) * K + k0 + ac;
            const unsigned ldsoff = (unsigned)(size_t)&As[buf][ar][ac];
            asm volatile("global_load_async_to_lds_b128 %0, %1, off"
                         :: "v"(ldsoff), "v"((unsigned long long)(size_t)gp)
                         : "memory");
        }
    };
    auto loadB_regs = [&](int k0, uint4* br) {
#pragma unroll
        for (int j = 0; j < 2; ++j) {               // 64x64 u16 = 512 x 16B chunks
            const int id = t + j * 256;
            const int kr = id >> 3, nc = (id & 7) * 8;
            br[j] = *(const uint4*)(B + (size_t)(k0 + kr) * N + gn0 + nc);
        }
    };
    auto storeB_lds = [&](const uint4* br, int buf) {
#pragma unroll
        for (int j = 0; j < 2; ++j) {
            const int id = t + j * 256;
            const int kr = id >> 3, nc = (id & 7) * 8;
            u16 tmp[8];
            *(uint4*)tmp = br[j];
#pragma unroll
            for (int e = 0; e < 8; ++e) Bst[buf][nc + e][kr] = tmp[e];
        }
    };
    auto compute = [&](int buf) {
#pragma unroll
        for (int kc = 0; kc < 2; ++kc) {            // two 32-wide K chunks
            FragA a0, a1, b0, b1;
            const int kc32 = kc * 32;
            a0.q[0] = *(const uint4*)&As[buf][mi2 * 32 + row16][kc32 + kb8];
            a0.q[1] = *(const uint4*)&As[buf][mi2 * 32 + row16][kc32 + 16 + kb8];
            a1.q[0] = *(const uint4*)&As[buf][mi2 * 32 + 16 + row16][kc32 + kb8];
            a1.q[1] = *(const uint4*)&As[buf][mi2 * 32 + 16 + row16][kc32 + 16 + kb8];
            b0.q[0] = *(const uint4*)&Bst[buf][nj0 * 16 + row16][kc32 + kb8];
            b0.q[1] = *(const uint4*)&Bst[buf][nj0 * 16 + row16][kc32 + 16 + kb8];
            b1.q[0] = *(const uint4*)&Bst[buf][(nj0 + 1) * 16 + row16][kc32 + kb8];
            b1.q[1] = *(const uint4*)&Bst[buf][(nj0 + 1) * 16 + row16][kc32 + 16 + kb8];
            acc[0][0] = wmma_bf16(a0, b0, acc[0][0]);
            acc[0][1] = wmma_bf16(a0, b1, acc[0][1]);
            acc[1][0] = wmma_bf16(a1, b0, acc[1][0]);
            acc[1][1] = wmma_bf16(a1, b1, acc[1][1]);
        }
    };

    // ---- software pipeline over K (BK = 64) ----
    uint4 br[2];
    stageA_async(0, 0);
    loadB_regs(0, br);
    storeB_lds(br, 0);
    asm volatile("s_wait_asynccnt 0" ::: "memory");
    __syncthreads();

    for (int k0 = 0; k0 < K; k0 += 64) {
        const int cur  = (k0 >> 6) & 1;
        const int nxt  = cur ^ 1;
        const bool more = (k0 + 64 < K);
        if (more) {
            stageA_async(k0 + 64, nxt);   // async DMA into other buffer
            loadB_regs(k0 + 64, br);      // global loads in flight during compute
        }
        compute(cur);                     // 8 WMMAs
        if (more) {
            storeB_lds(br, nxt);
            asm volatile("s_wait_asynccnt 0" ::: "memory");
        }
        __syncthreads();
    }

    // ---- store ----
#pragma unroll
    for (int im = 0; im < 2; ++im) {
#pragma unroll
        for (int in = 0; in < 2; ++in) {
#pragma unroll
            for (int r = 0; r < 8; ++r) {
                const int gm = gm0 + mi2 * 32 + im * 16 + r + 8 * half;
                const int gn = gn0 + (nj0 + in) * 16 + row16;
                const float vv = acc[im][in][r];
                if (mode == 0) {
                    ((u16*)Cout)[(size_t)gm * N + gn] = f32_to_bf16(vv);
                } else if (mode == 1) {
                    ((u16*)Cout)[(size_t)gn * M + gm] = f32_to_bf16(vv);  // [N][M]
                } else {
                    ((float*)Cout)[(size_t)gm * N + gn] = vv;
                }
            }
        }
    }
}

// ---------------- fused attention (online stats + emit pass) ----------------
// Qp,Kp bf16 [4096][1024] (col = h*64+kd); Vt bf16 [1024][4096] (row = h*64+dv)
// Writes: attnP f32 [B][H][S][S] (normalized), AttnO bf16 [4096][1024]
__global__ __launch_bounds__(256)
void k_attention(const u16* __restrict__ Qp, const u16* __restrict__ Kp,
                 const u16* __restrict__ Vt, const int* __restrict__ mask,
                 u16* __restrict__ AttnO, float* __restrict__ attnP) {
    __shared__ __align__(16) u16 ptile[8][16][40];   // per-wave P tile 16x32 bf16

    const int lane  = threadIdx.x & 31;
    const int wave  = threadIdx.x >> 5;
    const int tile  = blockIdx.x * 8 + wave;          // 4096 tiles total
    const int qt    = tile & 127;
    const int h     = (tile >> 7) & 15;
    const int b     = tile >> 11;
    const int qbase = qt * 16;
    const int row16 = lane & 15;
    const int half  = lane >> 4;
    const int kb8   = half * 8;

    // Q A-fragments (16x64 = 2 chunks of 16x32)
    FragA qa0, qa1;
    {
        const u16* qrow = Qp + (size_t)(b * SS + qbase + row16) * DM + h * DK;
        qa0.q[0] = *(const uint4*)(qrow + kb8);
        qa0.q[1] = *(const uint4*)(qrow + 16 + kb8);
        qa1.q[0] = *(const uint4*)(qrow + 32 + kb8);
        qa1.q[1] = *(const uint4*)(qrow + 48 + kb8);
    }
    const int* mrow = mask + b * SS;

    // S-tile pair (32 keys) recompute helper
    auto computeS = [&](int kb, f32x8& s0, f32x8& s1) {
#pragma unroll
        for (int sub = 0; sub < 2; ++sub) {
            const int key = kb + sub * 16 + row16;
            const u16* krow = Kp + (size_t)(b * SS + key) * DM + h * DK;
            FragA kf0, kf1;
            kf0.q[0] = *(const uint4*)(krow + kb8);
            kf0.q[1] = *(const uint4*)(krow + 16 + kb8);
            kf1.q[0] = *(const uint4*)(krow + 32 + kb8);
            kf1.q[1] = *(const uint4*)(krow + 48 + kb8);
            f32x8 c = zero8();
            c = wmma_bf16(qa0, kf0, c);
            c = wmma_bf16(qa1, kf1, c);
            const bool msk = (mrow[key] == 0);
            f32x8 s;
#pragma unroll
            for (int r = 0; r < 8; ++r) s[r] = msk ? NEGV : c[r] * SCALE;
            if (sub == 0) s0 = s; else s1 = s;
        }
    };

    // ---- pass 1: online row max + rescaled exp-sum ----
    float m[8], l[8];
#pragma unroll
    for (int r = 0; r < 8; ++r) { m[r] = -1e30f; l[r] = 0.0f; }

    for (int kb = 0; kb < SS; kb += 32) {
        f32x8 s0, s1;
        computeS(kb, s0, s1);
#pragma unroll
        for (int r = 0; r < 8; ++r) {
            const float rm = redmax16(fmaxf(s0[r], s1[r]));
            const float mn = fmaxf(m[r], rm);
            const float e  = exp2f((s0[r] - mn) * L2E) + exp2f((s1[r] - mn) * L2E);
            l[r] = l[r] * exp2f((m[r] - mn) * L2E) + redsum16(e);
            m[r] = mn;
        }
    }
    float invl[8];
#pragma unroll
    for (int r = 0; r < 8; ++r) invl[r] = 1.0f / l[r];

    // ---- pass 2: emit normalized P, accumulate O = P*V ----
    f32x8 oc[4];
#pragma unroll
    for (int nc = 0; nc < 4; ++nc) oc[nc] = zero8();

    float* aprow = attnP + ((size_t)(b * NH + h) * SS + qbase) * SS;

    for (int kb = 0; kb < SS; kb += 32) {
        f32x8 s0, s1;
        computeS(kb, s0, s1);
#pragma unroll
        for (int sub = 0; sub < 2; ++sub) {
            const int key = kb + sub * 16 + row16;
#pragma unroll
            for (int r = 0; r < 8; ++r) {
                const int Mr = r + 8 * half;
                const float sv = (sub == 0) ? s0[r] : s1[r];
                const float p  = exp2f((sv - m[r]) * L2E) * invl[r];
                aprow[(size_t)Mr * SS + key] = p;                   // normalized prob
                ptile[wave][Mr][sub * 16 + row16] = f32_to_bf16(p); // stage for WMMA A
            }
        }
        asm volatile("s_wait_dscnt 0" ::: "memory");   // LDS cross-lane RAW inside wave

        FragA pf;
        pf.q[0] = *(const uint4*)&ptile[wave][row16][kb8];
        pf.q[1] = *(const uint4*)&ptile[wave][row16][16 + kb8];

#pragma unroll
        for (int nc = 0; nc < 4; ++nc) {
            const u16* vrow = Vt + (size_t)(h * DK + nc * 16 + row16) * MROWS + b * SS + kb;
            FragA vf;
            vf.q[0] = *(const uint4*)(vrow + kb8);
            vf.q[1] = *(const uint4*)(vrow + 16 + kb8);
            oc[nc] = wmma_bf16(pf, vf, oc[nc]);
        }
        asm volatile("s_wait_dscnt 0" ::: "memory");   // don't overwrite ptile early
    }

    // write O (head-concat layout, bf16)
#pragma unroll
    for (int nc = 0; nc < 4; ++nc) {
#pragma unroll
        for (int r = 0; r < 8; ++r) {
            const int grow = b * SS + qbase + r + 8 * half;
            const int gcol = h * DK + nc * 16 + row16;
            AttnO[(size_t)grow * DM + gcol] = f32_to_bf16(oc[nc][r]);
        }
    }
}

// ---------------- bias + residual + LayerNorm ----------------
__global__ __launch_bounds__(256)
void k_layernorm(const float* __restrict__ proj, const float* __restrict__ qin,
                 const float* __restrict__ bo, const float* __restrict__ gamma,
                 const float* __restrict__ beta, float* __restrict__ out) {
    __shared__ float sred[256];
    const int row = blockIdx.x;
    const int t   = threadIdx.x;

    float x[4];
    float lsum = 0.0f;
#pragma unroll
    for (int j = 0; j < 4; ++j) {
        const int i = t + j * 256;
        x[j] = proj[(size_t)row * DM + i] + bo[i] + qin[(size_t)row * DM + i];
        lsum += x[j];
    }
    sred[t] = lsum;
    __syncthreads();
    for (int s = 128; s > 0; s >>= 1) {
        if (t < s) sred[t] += sred[t + s];
        __syncthreads();
    }
    const float mean = sred[0] * (1.0f / DM);
    __syncthreads();

    float lvar = 0.0f;
#pragma unroll
    for (int j = 0; j < 4; ++j) { float d = x[j] - mean; lvar += d * d; }
    sred[t] = lvar;
    __syncthreads();
    for (int s = 128; s > 0; s >>= 1) {
        if (t < s) sred[t] += sred[t + s];
        __syncthreads();
    }
    const float rstd = rsqrtf(sred[0] * (1.0f / DM) + 1e-5f);

#pragma unroll
    for (int j = 0; j < 4; ++j) {
        const int i = t + j * 256;
        out[(size_t)row * DM + i] = (x[j] - mean) * rstd * gamma[i] + beta[i];
    }
}

// ---------------- host launch ----------------
extern "C" void kernel_launch(void* const* d_in, const int* in_sizes, int n_in,
                              void* d_out, int out_size, void* d_ws, size_t ws_size,
                              hipStream_t stream) {
    const float* q    = (const float*)d_in[0];
    const float* k    = (const float*)d_in[1];
    const float* v    = (const float*)d_in[2];
    const int*   mask = (const int*)d_in[3];
    const float* Wq   = (const float*)d_in[4];
    const float* Wk   = (const float*)d_in[5];
    const float* Wv   = (const float*)d_in[6];
    const float* Wo   = (const float*)d_in[7];
    const float* bo   = (const float*)d_in[8];
    const float* gam  = (const float*)d_in[9];
    const float* bet  = (const float*)d_in[10];

    float* out   = (float*)d_out;                         // [4096][1024]
    float* attnP = (float*)d_out + (size_t)MROWS * DM;    // [2][16][2048][2048]

    // workspace layout (bf16 buffers unless noted)
    char* ws = (char*)d_ws;
    const size_t ACT = (size_t)MROWS * DM * sizeof(u16);  // 8 MB
    const size_t WGT = (size_t)DM * DM * sizeof(u16);     // 2 MB
    u16*   qb   = (u16*)(ws);
    u16*   kb   = (u16*)(ws + ACT);
    u16*   vb   = (u16*)(ws + 2 * ACT);
    u16*   wqb  = (u16*)(ws + 3 * ACT);
    u16*   wkb  = (u16*)(ws + 3 * ACT + WGT);
    u16*   wvb  = (u16*)(ws + 3 * ACT + 2 * WGT);
    u16*   wob  = (u16*)(ws + 3 * ACT + 3 * WGT);
    u16*   Qp   = (u16*)(ws + 3 * ACT + 4 * WGT);
    u16*   Kp   = (u16*)(ws + 4 * ACT + 4 * WGT);
    u16*   Vt   = (u16*)(ws + 5 * ACT + 4 * WGT);         // transposed [1024][4096]
    u16*   Oat  = (u16*)(ws + 6 * ACT + 4 * WGT);
    float* proj = (float*)(ws + 7 * ACT + 4 * WGT);       // f32 [4096][1024]

    const int NACT = MROWS * DM, NWGT = DM * DM;

    // 1) f32 -> bf16
    k_f2bf<<<(NACT + 255) / 256, 256, 0, stream>>>(q,  qb,  NACT);
    k_f2bf<<<(NACT + 255) / 256, 256, 0, stream>>>(k,  kb,  NACT);
    k_f2bf<<<(NACT + 255) / 256, 256, 0, stream>>>(v,  vb,  NACT);
    k_f2bf<<<(NWGT + 255) / 256, 256, 0, stream>>>(Wq, wqb, NWGT);
    k_f2bf<<<(NWGT + 255) / 256, 256, 0, stream>>>(Wk, wkb, NWGT);
    k_f2bf<<<(NWGT + 255) / 256, 256, 0, stream>>>(Wv, wvb, NWGT);
    k_f2bf<<<(NWGT + 255) / 256, 256, 0, stream>>>(Wo, wob, NWGT);

    // 2) projections (WMMA GEMMs); V written transposed for the PV B-operand
    dim3 gg(MROWS / 128, DM / 64);
    k_gemm_bf16<<<gg, 256, 0, stream>>>(qb, wqb, Qp, MROWS, DM, DM, 0);
    k_gemm_bf16<<<gg, 256, 0, stream>>>(kb, wkb, Kp, MROWS, DM, DM, 0);
    k_gemm_bf16<<<gg, 256, 0, stream>>>(vb, wvb, Vt, MROWS, DM, DM, 1);

    // 3) fused attention: 4096 query tiles, 8 waves/block
    k_attention<<<512, 256, 0, stream>>>(Qp, Kp, Vt, mask, Oat, attnP);

    // 4) output projection (f32 result)
    k_gemm_bf16<<<gg, 256, 0, stream>>>(Oat, wob, proj, MROWS, DM, DM, 2);

    // 5) bias + residual + LayerNorm
    k_layernorm<<<MROWS, 256, 0, stream>>>(proj, q, bo, gam, bet, out);
}
